// FAPBlur_89008902242780
// MI455X (gfx1250) — matrix operand order repaired and checked by reading, not compiled, not run on hardware
//
#include <hip/hip_runtime.h>
#include <hip/hip_bf16.h>

typedef __attribute__((ext_vector_type(16))) _Float16 v16h;
typedef __attribute__((ext_vector_type(8)))  float    v8f;

#define QBINS   16
#define TILE_W  16
#define TILE_H  128
#define RMAX    34
#define XS_VROWS (TILE_H + 2*RMAX)  // 196 valid halo rows
#define XS_VCOLS (TILE_W + 2*RMAX)  // 84 valid halo cols
#define XS_ROWS  212                // padded rows (zero-filled pad)
#define XS_STRIDE 132               // padded cols; bank-spreading stride
#define HS_STRIDE 216               // transposed Hs: [16 cols][216 rows]
#define HS_ROWS  208                // 13 chunks * 16 rows max
#define TAP_OFF 64                  // zero-padded tap table offset
#define TAP_LEN 192
#define IMG_W   1024
#define IMG_H   1024

__device__ __forceinline__ int pixel_bin(float py, float px, const int* f) {
    float dmin = 1e30f;
#pragma unroll
    for (int j = 0; j < 3; ++j) {
        float dy = py - (float)f[2*j];
        float dx = px - (float)f[2*j + 1];
        float d2 = dy*dy + dx*dx;
        dmin = fminf(dmin, d2);
    }
    float d   = sqrtf(dmin);
    float z   = 1.5f * (d * 0.07f - 18.2f);
    float sig = 0.84f / (1.0f + __expf(-z));          // sigma_mm
    float spx = fminf(fmaxf(sig * (1.0f/0.07f), 0.0f), 12.0f);
    int b = (int)ceilf(spx * (1.0f/0.75f)) - 1;       // searchsorted(left) - 1
    return min(max(b, 0), QBINS - 1);
}

__global__ __launch_bounds__(256)
void fap_blur_wmma(const float* __restrict__ xin,
                   const int*   __restrict__ fix,
                   float*       __restrict__ out)
{
    __shared__ _Float16 Xs[XS_ROWS * XS_STRIDE];      // f16 halo, zero-padded
    __shared__ _Float16 Hs[TILE_W * HS_STRIDE];       // transposed h-strip
    __shared__ float    tapsF[80];
    __shared__ _Float16 tapsH[TAP_LEN];               // zero-padded band table
    __shared__ unsigned char binB[TILE_H * TILE_W];
    __shared__ unsigned int  binMask;
    __shared__ float         tapSum;

    const int tid  = threadIdx.x;
    const int lane = tid & 31;
    const int wid  = tid >> 5;

    // block -> (plane, tile)
    int bid   = blockIdx.x;
    int tx    = bid & 63;            // 64 x-tiles
    int ty    = (bid >> 6) & 7;      // 8 y-tiles
    int plane = bid >> 9;            // 24 planes (B*C)
    int batch = plane / 3;
    int x0 = tx * TILE_W;
    int y0 = ty * TILE_H;

    const float* xp = xin + (size_t)plane * IMG_H * IMG_W;
    float*       op = out + (size_t)plane * IMG_H * IMG_W;

    int f[6];
#pragma unroll
    for (int j = 0; j < 6; ++j) f[j] = fix[batch * 6 + j];

    if (tid == 0) binMask = 0u;

    // Zero-fill Hs once: V-pass may read a few rows past what H-pass wrote
    // (they meet zero band taps, but WMMA propagates NaN through 0-weight
    // products, so the bits must be finite).
    for (int i = tid; i < TILE_W * HS_STRIDE; i += 256)
        Hs[i] = (_Float16)0.f;
    __syncthreads();

    // ---- stage max-halo tile into LDS as f16; zero-fill pad region + image
    //      border (matches the reference's zero-pad convolution) ----
    for (int i = tid; i < XS_ROWS * XS_STRIDE; i += 256) {
        int rr = i / XS_STRIDE, cc = i - rr * XS_STRIDE;
        float v = 0.0f;
        if (rr < XS_VROWS && cc < XS_VCOLS) {
            int gy = y0 - RMAX + rr;
            int gx = x0 - RMAX + cc;
            if ((unsigned)gy < (unsigned)IMG_H && (unsigned)gx < (unsigned)IMG_W)
                v = xp[gy * IMG_W + gx];
        }
        Xs[i] = (_Float16)v;
    }

    // ---- per-pixel bins + present-bin bitmask ----
    unsigned int lm = 0u;
    for (int i = tid; i < TILE_H * TILE_W; i += 256) {
        int py = i >> 4, px = i & 15;
        int b = pixel_bin((float)(y0 + py), (float)(x0 + px), f);
        binB[i] = (unsigned char)b;
        lm |= (1u << b);
    }
    atomicOr(&binMask, lm);
    __syncthreads();
    const unsigned int mask = binMask;

    const int m  = lane & 15;              // A-row for this lane
    const int kb = (lane >> 4) << 3;       // A K-subblock base / 2 (base 0 or 16)
    const int n  = lane & 15;              // B/C/D column for this lane

    for (int q = 0; q < QBINS; ++q) {      // uniform loop over present bins
        if (!(mask & (1u << q))) continue;

        float s    = 0.375f + 0.75f * (float)q;
        int   r    = (int)(3.0f * s);      // ref: int(3*s)
        int   ks   = 2*r + 1;
        int   SH   = TILE_H + 2*r;         // h-strip height
        int   Kdim = TILE_W + 2*r;         // GEMM K (16+2r)

        // ---- normalized Gaussian taps -> zero-padded f16 band table ----
        if (tid < ks) {
            float a = (float)(tid - r);
            tapsF[tid] = __expf(-0.5f * (a/s) * (a/s));
        }
        __syncthreads();
        if (tid == 0) {
            float ssum = 0.f;
            for (int t = 0; t < ks; ++t) ssum += tapsF[t];
            tapSum = ssum;
        }
        __syncthreads();
        if (tid < TAP_LEN) {
            int t = tid - TAP_OFF;
            tapsH[tid] = (t >= 0 && t < ks) ? (_Float16)(tapsF[t] / tapSum)
                                            : (_Float16)0.f;
        }
        __syncthreads();

        const int nchunks = (SH + 15) >> 4;
        const int nk      = (Kdim + 31) >> 5;
        const int coff    = RMAX - r;      // shift into max-halo LDS tile

        // ================= H pass: H = X_halo * Gh (banded) =================
        // No guards: A elements with kk>=Kdim meet zero B taps (band), and
        // over-read rows/cols land in the zero-filled pad of Xs.
        for (int c = wid; c < nchunks; c += 8) {
            int row0 = c << 4;
            v8f acc = {};
            int abase = (row0 + m + coff) * XS_STRIDE + coff;
            for (int t = 0; t < nk; ++t) {
                int k0 = t << 5;
                v16h A;                    // contiguous k = 16h + s per lane half
#pragma unroll
                for (int i = 0; i < 8; ++i) {
                    int k1 = 2*kb + 2*i;
                    int kk = k0 + k1;
                    A[2*i]   = Xs[abase + kk];
                    A[2*i+1] = Xs[abase + kk + 1];
                }
                v16h B;                    // B[kk][n] = g[kk-n], zeros from table
                int tb = TAP_OFF + k0 + ((lane >> 4) << 4) - n;
#pragma unroll
                for (int i = 0; i < 8; ++i) {
                    B[2*i]   = tapsH[tb + 2*i];
                    B[2*i+1] = tapsH[tb + 2*i + 1];
                }
                acc = __builtin_amdgcn_wmma_f32_16x16x32_f16(
                        false, A, false, B, (short)0, acc, false, false);
            }
            // transposed store: 8 contiguous halfs per lane (b128-mergeable)
            int Mb = (lane >> 4) << 3;     // C/D layout: M = i (+8 for hi lanes)
            int hb = n * HS_STRIDE + row0 + Mb;
#pragma unroll
            for (int i = 0; i < 8; ++i)
                Hs[hb + i] = (_Float16)acc[i];
        }
        __syncthreads();

        // ============== V pass: Out = Gv (banded) * H_strip ==============
        {
            int ob = wid << 4;             // one 16-row output chunk per wave
            v8f acc = {};
            for (int t = 0; t < nk; ++t) {
                int k0 = t << 5;
                v16h A;                    // A[m][kk] = g[kk-m], zeros from table
                int ta = TAP_OFF + k0 - m;
#pragma unroll
                for (int i = 0; i < 8; ++i) {
                    int k1 = 2*kb + 2*i;
                    A[2*i]   = tapsH[ta + k1];
                    A[2*i+1] = tapsH[ta + k1 + 1];
                }
                v16h B;                    // B[kk][n] = Hs[n][ob+kk] (contiguous!)
                int hb = n * HS_STRIDE + ob + k0 + ((lane >> 4) << 4);
#pragma unroll
                for (int i = 0; i < 8; ++i) {
                    B[2*i]   = Hs[hb + 2*i];
                    B[2*i+1] = Hs[hb + 2*i + 1];
                }
                acc = __builtin_amdgcn_wmma_f32_16x16x32_f16(
                        false, A, false, B, (short)0, acc, false, false);
            }
            // per-pixel bin-match select + store (each pixel hit by exactly one q)
            int Mb = (lane >> 4) << 3;
#pragma unroll
            for (int i = 0; i < 8; ++i) {
                int py = ob + Mb + i;
                if ((int)binB[py * TILE_W + n] == q)
                    op[(size_t)(y0 + py) * IMG_W + (x0 + n)] = acc[i];
            }
        }
        __syncthreads();
    }
}

extern "C" void kernel_launch(void* const* d_in, const int* in_sizes, int n_in,
                              void* d_out, int out_size, void* d_ws, size_t ws_size,
                              hipStream_t stream) {
    const float* x   = (const float*)d_in[0];   // [8,3,1024,1024] f32
    const int*   fix = (const int*)d_in[1];     // [8,3,2] i32
    float*       out = (float*)d_out;           // [8,3,1024,1024] f32

    dim3 grid(24 * 64 * 8);   // planes * x-tiles * y-tiles
    dim3 block(256);          // 8 wave32s
    hipLaunchKernelGGL(fap_blur_wmma, grid, block, 0, stream, x, fix, out);
}
